// GATNet_8667244003465
// MI455X (gfx1250) — matrix operand order repaired and checked by reading, not compiled, not run on hardware
//
#include <hip/hip_runtime.h>
#include <hip/hip_bf16.h>
#include <stdint.h>

// ---------------- problem constants (from reference) ----------------
#define B_    2
#define NX_   20000     // x nodes
#define F_    128       // input features (= K of GEMM1)
#define NS_   16000     // gathered source nodes (n_id1)
#define ND1_  8000
#define ND2_  4000
#define E1_   100000
#define E2_   60000
#define H_    12        // heads layer 1
#define CH_   128       // channels per head
#define NN_   1536      // H_*CH_ (= N of GEMM1)

#define MT_   2000      // (B_*NS_)/16  M-tiles
#define NT_   96        // NN_/16       N-tiles
#define KT_   4         // F_/32        K-steps of 32

typedef __attribute__((ext_vector_type(16))) __bf16 v16bf;
typedef __attribute__((ext_vector_type(8)))  float  v8f;
typedef __attribute__((ext_vector_type(4)))  int    v4i;

#if defined(__has_builtin)
#if __has_builtin(__builtin_amdgcn_global_load_async_to_lds_b128)
#define HAVE_ASYNC_LDS 1
#endif
#endif

__device__ __forceinline__ unsigned short f2bf(float f) {
    union { float f; unsigned u; } v; v.f = f;
    unsigned u = v.u + 0x7fffu + ((v.u >> 16) & 1u);   // round-to-nearest-even
    return (unsigned short)(u >> 16);
}

// ---- K0a: swizzle W1 (f32 [128,1536]) into bf16 B-fragment layout ----
// Layout: [nt][kt][lane][16 bf16]; lane<16: N=nt*16+lane, VGPR v: K=kt*32+2v(+1)
//         lane>=16: same N set, K offset +16.
__global__ __launch_bounds__(256) void k_swzW(const float* __restrict__ W1,
                                              unsigned* __restrict__ wsw) {
    int t = blockIdx.x * 256 + threadIdx.x;
    if (t >= NT_ * KT_ * 32 * 8) return;
    int v = t & 7, lane = (t >> 3) & 31, kt = (t >> 8) & 3, nt = t >> 10;
    int nl = lane & 15, hf = lane >> 4;
    int ncol = nt * 16 + nl;
    int k = kt * 32 + 2 * v + 16 * hf;
    unsigned short l0 = f2bf(W1[(size_t)k * NN_ + ncol]);
    unsigned short l1 = f2bf(W1[(size_t)(k + 1) * NN_ + ncol]);
    wsw[t] = ((unsigned)l1 << 16) | l0;
}

// ---- K0b: gather x rows via n_id1 and swizzle into bf16 A-fragment layout ----
// A 16x32 bf16 layout: lane<16: M=lane, VGPR v: K=2v(+1) (same k order as B);
//                      lane>=16: K offset +16.
__global__ __launch_bounds__(256) void k_swzA(const float* __restrict__ x,
                                              const int* __restrict__ n_id1,
                                              unsigned* __restrict__ asw) {
    int t = blockIdx.x * 256 + threadIdx.x;
    if (t >= MT_ * KT_ * 32 * 8) return;
    int v = t & 7, lane = (t >> 3) & 31, kt = (t >> 8) & 3, mt = t >> 10;
    int m = lane & 15, hf = lane >> 4;
    int row = mt * 16 + m;                 // 0..31999 (b-major over B_*NS_)
    int b = row / NS_;
    int n = n_id1[row % NS_];
    int kb = kt * 32 + 2 * v + 16 * hf;
    const float* xr = x + ((size_t)b * NX_ + n) * F_ + kb;
    unsigned short l0 = f2bf(xr[0]);
    unsigned short l1 = f2bf(xr[1]);
    asw[t] = ((unsigned)l1 << 16) | l0;
}

// ---- K1: WMMA GEMM  hs[32000,1536] = A(bf16) x W1(bf16), f32 accumulate ----
// Block = 8 waves, all sharing one M-tile (mt = blockIdx.x/3, ng = 8*(blockIdx.x%3)+wave).
// A fragments (4 KB) staged once per block into LDS via global_load_async_to_lds_b128;
// B fragments stream from L2 (Wsw is 384 KB, L2-resident, no intra-block reuse).
__global__ __launch_bounds__(256) void k_gemm1(const unsigned* __restrict__ Araw,
                                               const v16bf* __restrict__ Bm,
                                               float* __restrict__ hs) {
    __shared__ __align__(16) unsigned lds_a[1024];   // 4 KB: [kt][lane] 32B fragments
    int mt = blockIdx.x / 3;
    int ng = (blockIdx.x % 3) * 8 + (threadIdx.x >> 5);
    int lane = threadIdx.x & 31;

    // stage this M-tile's A fragments into LDS (each thread moves 16 B)
    const char* gsrc = (const char*)Araw + (size_t)mt * 4096 + (size_t)threadIdx.x * 16;
#if defined(HAVE_ASYNC_LDS)
    __builtin_amdgcn_global_load_async_to_lds_b128(
        (v4i*)gsrc,
        (v4i*)((char*)lds_a + threadIdx.x * 16),
        0, 0);
#if __has_builtin(__builtin_amdgcn_s_wait_asynccnt)
    __builtin_amdgcn_s_wait_asynccnt(0);
#else
    asm volatile("s_wait_asynccnt 0x0" ::: "memory");
#endif
#else
    ((uint4*)lds_a)[threadIdx.x] = *(const uint4*)gsrc;
#endif
    __syncthreads();

    const v16bf* lA = (const v16bf*)lds_a;
    v8f c0 = {}, c1 = {}, c2 = {}, c3 = {};
#pragma unroll
    for (int kt = 0; kt < KT_; ++kt) {
        v16bf a  = lA[kt * 32 + lane];
        v16bf b0 = Bm[((ng * 4 + 0) * KT_ + kt) * 32 + lane];
        v16bf b1 = Bm[((ng * 4 + 1) * KT_ + kt) * 32 + lane];
        v16bf b2 = Bm[((ng * 4 + 2) * KT_ + kt) * 32 + lane];
        v16bf b3 = Bm[((ng * 4 + 3) * KT_ + kt) * 32 + lane];
        c0 = __builtin_amdgcn_wmma_f32_16x16x32_bf16(false, a, false, b0, (short)0, c0, false, false);
        c1 = __builtin_amdgcn_wmma_f32_16x16x32_bf16(false, a, false, b1, (short)0, c1, false, false);
        c2 = __builtin_amdgcn_wmma_f32_16x16x32_bf16(false, a, false, b2, (short)0, c2, false, false);
        c3 = __builtin_amdgcn_wmma_f32_16x16x32_bf16(false, a, false, b3, (short)0, c3, false, false);
    }
    // C layout: VGPR r, lane<16: M=r, N=lane; lane>=16: M=8+r, N=lane-16
    int nl = lane & 15;
    int rbase = mt * 16 + (lane >> 4) * 8;
    int ncol = ng * 64 + nl;
#pragma unroll
    for (int r = 0; r < 8; ++r) {
        size_t off = (size_t)(rbase + r) * NN_ + ncol;
        hs[off +  0] = c0[r];
        hs[off + 16] = c1[r];
        hs[off + 32] = c2[r];
        hs[off + 48] = c3[r];
    }
}

// ---- K2: per-(row,head) attention logits: es = h.a_src, ed = h.a_dst ----
__global__ __launch_bounds__(256) void k_attn(const float* __restrict__ hs,
                                              const float* __restrict__ asrc,
                                              const float* __restrict__ adst,
                                              float* __restrict__ es,
                                              float* __restrict__ ed) {
    int wid  = (blockIdx.x * 256 + threadIdx.x) >> 5;
    int lane = threadIdx.x & 31;
    int row = wid / H_, h = wid % H_;
    if (row >= B_ * NS_) return;
    const float* hr = hs + (size_t)row * NN_ + h * CH_;
    const float* as = asrc + h * CH_;
    const float* ad = adst + h * CH_;
    float s = 0.f, d = 0.f;
#pragma unroll
    for (int j = 0; j < 4; ++j) {
        float v = hr[lane + 32 * j];
        s += v * as[lane + 32 * j];
        d += v * ad[lane + 32 * j];
    }
#pragma unroll
    for (int o = 16; o > 0; o >>= 1) {
        s += __shfl_xor(s, o, 32);
        d += __shfl_xor(d, o, 32);
    }
    if (lane == 0) { es[row * H_ + h] = s; ed[row * H_ + h] = d; }
}

// ---- K3: layer-1 edges: ex = exp(lrelu(es[src]+ed[rid(dst)])); num/den atomics ----
__global__ __launch_bounds__(256) void k_edge1(const int* __restrict__ src1,
                                               const int* __restrict__ dst1,
                                               const int* __restrict__ rn1,
                                               const float* __restrict__ es,
                                               const float* __restrict__ ed,
                                               const float* __restrict__ hs,
                                               float* __restrict__ den1,
                                               float* __restrict__ num1) {
    __shared__ float exsh[H_];
    int e = blockIdx.x, b = blockIdx.y;
    int t = threadIdx.x;
    int s = src1[e], d = dst1[e];
    if (t < H_) {
        int rid = rn1[d];
        float a = es[(size_t)(b * NS_ + s) * H_ + t] + ed[(size_t)(b * NS_ + rid) * H_ + t];
        a = a > 0.f ? a : 0.2f * a;                 // leaky_relu(0.2)
        float ex = __expf(a);                       // softmax shift-invariant: skip max
        exsh[t] = ex;
        atomicAdd(&den1[(size_t)(b * ND1_ + d) * H_ + t], ex);
    }
    __syncthreads();
    const float* hr = hs + ((size_t)b * NS_ + s) * NN_;
    float* nr = num1 + ((size_t)b * ND1_ + d) * NN_;
    for (int idx = t; idx < NN_; idx += 256)
        atomicAdd(&nr[idx], exsh[idx >> 7] * hr[idx]);
}

// ---- K4: out1 = relu(num/den + bias1); hs2 = out1 . W2 (fused projection) ----
__global__ __launch_bounds__(256) void k_reluproj(const float* __restrict__ num1,
                                                  const float* __restrict__ den1,
                                                  const float* __restrict__ bias1,
                                                  const float* __restrict__ W2,
                                                  float* __restrict__ hs2) {
    __shared__ float red[256];
    int row = blockIdx.x;                  // b*ND1_+i
    int t = threadIdx.x;
    const float* nr = num1 + (size_t)row * NN_;
    const float* dr = den1 + (size_t)row * H_;
    float acc = 0.f;
    for (int idx = t; idx < NN_; idx += 256) {
        float v = nr[idx] / (dr[idx >> 7] + 1e-16f) + bias1[idx];
        v = v > 0.f ? v : 0.f;
        acc += v * W2[idx];
    }
    red[t] = acc; __syncthreads();
#pragma unroll
    for (int o = 128; o > 0; o >>= 1) {
        if (t < o) red[t] += red[t + o];
        __syncthreads();
    }
    if (t == 0) hs2[row] = red[0];
}

// ---- K5: layer-2 edges (heads=1, ch=1) ----
__global__ __launch_bounds__(256) void k_edge2(const int* __restrict__ src2,
                                               const int* __restrict__ dst2,
                                               const int* __restrict__ rn2,
                                               const float* __restrict__ hs2,
                                               const float* __restrict__ as2,
                                               const float* __restrict__ ad2,
                                               float* __restrict__ den2,
                                               float* __restrict__ num2) {
    int tid = blockIdx.x * 256 + threadIdx.x;
    if (tid >= E2_ * B_) return;
    int e = tid % E2_, b = tid / E2_;
    int s = src2[e], d = dst2[e], rid = rn2[d];
    float hsv = hs2[b * ND1_ + s];
    float hdv = hs2[b * ND1_ + rid];
    float a = hsv * as2[0] + hdv * ad2[0];
    a = a > 0.f ? a : 0.2f * a;
    float ex = __expf(a);
    atomicAdd(&den2[b * ND2_ + d], ex);
    atomicAdd(&num2[b * ND2_ + d], ex * hsv);
}

// ---- K6: finalize ----
__global__ __launch_bounds__(256) void k_final(const float* __restrict__ num2,
                                               const float* __restrict__ den2,
                                               const float* __restrict__ bias2,
                                               float* __restrict__ out) {
    int tid = blockIdx.x * 256 + threadIdx.x;
    if (tid >= B_ * ND2_) return;
    out[tid] = num2[tid] / (den2[tid] + 1e-16f) + bias2[0];
}

extern "C" void kernel_launch(void* const* d_in, const int* in_sizes, int n_in,
                              void* d_out, int out_size, void* d_ws, size_t ws_size,
                              hipStream_t stream) {
    const float* x    = (const float*)d_in[0];
    const int*   n_id1= (const int*)  d_in[1];
    const int*   rn1  = (const int*)  d_in[2];
    const int*   src1 = (const int*)  d_in[3];
    const int*   dst1 = (const int*)  d_in[4];
    const int*   rn2  = (const int*)  d_in[5];
    const int*   src2 = (const int*)  d_in[6];
    const int*   dst2 = (const int*)  d_in[7];
    const float* W1   = (const float*)d_in[8];
    const float* as1  = (const float*)d_in[9];
    const float* ad1  = (const float*)d_in[10];
    const float* b1   = (const float*)d_in[11];
    const float* W2   = (const float*)d_in[12];
    const float* as2  = (const float*)d_in[13];
    const float* ad2  = (const float*)d_in[14];
    const float* b2   = (const float*)d_in[15];

    // workspace carve-up (256B aligned)
    char* ws = (char*)d_ws;
    size_t off = 0;
    auto carve = [&](size_t bytes) -> char* {
        char* p = ws + off;
        off = (off + bytes + 255) & ~(size_t)255;
        return p;
    };
    float*    hs   = (float*)   carve((size_t)B_ * NS_ * NN_ * 4);   // 196.6 MB
    float*    num1 = (float*)   carve((size_t)B_ * ND1_ * NN_ * 4);  //  98.3 MB
    unsigned* asw  = (unsigned*)carve((size_t)MT_ * KT_ * 32 * 8 * 4);
    unsigned* wsw  = (unsigned*)carve((size_t)NT_ * KT_ * 32 * 8 * 4);
    float*    es   = (float*)   carve((size_t)B_ * NS_ * H_ * 4);
    float*    ed   = (float*)   carve((size_t)B_ * NS_ * H_ * 4);
    float*    den1 = (float*)   carve((size_t)B_ * ND1_ * H_ * 4);
    float*    hs2  = (float*)   carve((size_t)B_ * ND1_ * 4);
    float*    den2 = (float*)   carve((size_t)B_ * ND2_ * 4);
    float*    num2 = (float*)   carve((size_t)B_ * ND2_ * 4);

    (void)hipMemsetAsync(num1, 0, (size_t)B_ * ND1_ * NN_ * 4, stream);
    (void)hipMemsetAsync(den1, 0, (size_t)B_ * ND1_ * H_ * 4, stream);
    (void)hipMemsetAsync(den2, 0, (size_t)B_ * ND2_ * 4, stream);
    (void)hipMemsetAsync(num2, 0, (size_t)B_ * ND2_ * 4, stream);

    k_swzW<<<(NT_ * KT_ * 32 * 8 + 255) / 256, 256, 0, stream>>>(W1, wsw);
    k_swzA<<<(MT_ * KT_ * 32 * 8 + 255) / 256, 256, 0, stream>>>(x, n_id1, asw);
    // 6000 blocks x 8 waves = 2000 mtiles * 24 n-groups (3 blocks per mtile)
    k_gemm1<<<6000, 256, 0, stream>>>(asw, (const v16bf*)wsw, hs);
    // 384000 waves = 32000 rows * 12 heads; 8 waves/block
    k_attn<<<48000, 256, 0, stream>>>(hs, as1, ad1, es, ed);
    k_edge1<<<dim3(E1_, B_), 256, 0, stream>>>(src1, dst1, rn1, es, ed, hs, den1, num1);
    k_reluproj<<<B_ * ND1_, 256, 0, stream>>>(num1, den1, b1, W2, hs2);
    k_edge2<<<(E2_ * B_ + 255) / 256, 256, 0, stream>>>(src2, dst2, rn2, hs2, as2, ad2, den2, num2);
    k_final<<<(B_ * ND2_ + 255) / 256, 256, 0, stream>>>(num2, den2, b2, (float*)d_out);
    (void)in_sizes; (void)n_in; (void)out_size; (void)ws_size;
}
